// GraphElectronModel2_43928925503631
// MI455X (gfx1250) — matrix operand order repaired and checked by reading, not compiled, not run on hardware
//
#include <hip/hip_runtime.h>
#include <hip/hip_bf16.h>

typedef __attribute__((ext_vector_type(16))) _Float16 v16h;
typedef __attribute__((ext_vector_type(8)))  _Float16 v8h;
typedef __attribute__((ext_vector_type(8)))  float    v8f;

#define D_FEAT 128
#define HID    64
#define CH     128

// ---------------------------------------------------------------------------
// Weight prep: f32 [K,N] row-major -> f16 WMMA B-fragments.
// Fragment storage: out[((kt*NT + nt)*32 + lane)*16 + h]
// B layout (16-bit, 32x16 tile): lanes 0-15 hold K = kt*32 + 0..7, 16..23 (halves),
//                                lanes 16-31 hold K = kt*32 + 8..15, 24..31.
// Column n = nt*16 + (lane & 15).
// ---------------------------------------------------------------------------
__global__ void prep_w16(const float* __restrict__ W, _Float16* __restrict__ out,
                         int K, int N) {
  int tid = blockIdx.x * blockDim.x + threadIdx.x;
  if (tid >= K * N) return;
  int h    = tid & 15;
  int lane = (tid >> 4) & 31;
  int fid  = tid >> 9;
  int NT   = N >> 4;
  int nt   = fid % NT;
  int kt   = fid / NT;
  int k = kt * 32 + ((lane & 16) ? 8 : 0) + (h & 7) + ((h & 8) ? 16 : 0);
  int n = nt * 16 + (lane & 15);
  out[tid] = (_Float16)W[k * N + n];
}

__global__ void cvt_f32_f16(const float* __restrict__ in,
                            _Float16* __restrict__ out, int total) {
  int i = blockIdx.x * blockDim.x + threadIdx.x;
  if (i < total) out[i] = (_Float16)in[i];
}

__global__ void zero_f32(float* __restrict__ p, int total) {
  int i = blockIdx.x * blockDim.x + threadIdx.x;
  if (i < total) p[i] = 0.f;
}

// out_f16[i] = cnt>0 ? seg/cnt : 0   (CH == 128 -> i>>7)
__global__ void seg_mean_f16(const float* __restrict__ seg_sum,
                             const float* __restrict__ cnt,
                             _Float16* __restrict__ out, int total) {
  int i = blockIdx.x * blockDim.x + threadIdx.x;
  if (i >= total) return;
  float c = cnt[i >> 7];
  out[i] = (_Float16)((c > 0.f) ? seg_sum[i] / c : 0.f);
}

__device__ __forceinline__ v16h cat16(v8h lo, v8h hi) {
  return __builtin_shufflevector(lo, hi, 0, 1, 2, 3, 4, 5, 6, 7,
                                 8, 9, 10, 11, 12, 13, 14, 15);
}

// ---------------------------------------------------------------------------
// Edge MLP + scatter. One wave = one 32-edge tile (two 16-row M-subtiles that
// share every B fragment -> half the weight-fragment traffic per edge).
//   A fragments built straight from the f16 feature table (no cvt on hot path);
//   16-bit A striping: lanes 0-15: halves 0-7 -> K 0-7, halves 8-15 -> K 16-23;
//                      lanes 16-31: halves 0-7 -> K 8-15, halves 8-15 -> K 24-31.
//   GEMM1: [32,256]x[256,64] = 8 kt x 4 nt x 2 subtiles WMMA -> relu -> LDS(f16)
//   GEMM2: [32,64]x[64,128]  = 2 kt x 8 nt x 2 subtiles WMMA -> relu -> scatter
// ---------------------------------------------------------------------------
__global__ void __launch_bounds__(128)
edge_mlp_scatter(const _Float16* __restrict__ xh,
                 const int* __restrict__ senders,
                 const int* __restrict__ receivers,
                 const _Float16* __restrict__ W1h,  // K=256,N=64 fragments
                 const float* __restrict__ b1,
                 const _Float16* __restrict__ W2h,  // K=64,N=128 fragments
                 const float* __restrict__ b2,
                 float* __restrict__ seg_sum,
                 float* __restrict__ cnt,
                 int n_edges) {
  __shared__ __align__(32) _Float16 lds_h[4][32 * HID];  // 16 KB
  __shared__ int lds_ridx[4][32];

  const int wave   = threadIdx.x >> 5;
  const int lane   = threadIdx.x & 31;
  const int row    = lane & 15;
  const int hiHalf = (lane >> 4) & 1;
  const int mrow   = hiHalf * 8;
  const int off    = hiHalf * 8;

  const int maxTile = (n_edges + 31) >> 5;
  int tile = blockIdx.x * 4 + wave;
  const bool tileValid = (tile < maxTile);
  if (!tileValid) tile = maxTile - 1;
  const int ebase = tile << 5;

  // Edge indices for this lane's row in each subtile
  const int e0 = ebase + row;
  const int e1 = ebase + 16 + row;
  const bool v0 = tileValid && (e0 < n_edges);
  const bool v1 = tileValid && (e1 < n_edges);
  const int e0c = min(e0, n_edges - 1);
  const int e1c = min(e1, n_edges - 1);
  const int r0 = receivers[e0c], s0 = senders[e0c];
  const int r1 = receivers[e1c], s1 = senders[e1c];

  if (!hiHalf) {
    lds_ridx[wave][row]      = v0 ? r0 : -1;
    lds_ridx[wave][16 + row] = v1 ? r1 : -1;
  }

  // ---- GEMM1: [32,256] x [256,64] ----
  v8f acc1[2][4];
  #pragma unroll
  for (int t = 0; t < 2; ++t)
    #pragma unroll
    for (int nt = 0; nt < 4; ++nt) acc1[t][nt] = (v8f)0.0f;

  #pragma unroll
  for (int kt = 0; kt < 8; ++kt) {
    const _Float16* b0 = (kt < 4)
        ? (xh + (size_t)r0 * D_FEAT + kt * 32)
        : (xh + (size_t)s0 * D_FEAT + (kt - 4) * 32);
    const _Float16* b1p = (kt < 4)
        ? (xh + (size_t)r1 * D_FEAT + kt * 32)
        : (xh + (size_t)s1 * D_FEAT + (kt - 4) * 32);
    v16h a0 = cat16(*(const v8h*)(b0 + off),  *(const v8h*)(b0 + 16 + off));
    v16h a1 = cat16(*(const v8h*)(b1p + off), *(const v8h*)(b1p + 16 + off));
    #pragma unroll
    for (int nt = 0; nt < 4; ++nt) {
      v16h b = *(const v16h*)(W1h + (((kt << 2) + nt) << 9) + (lane << 4));
      acc1[0][nt] = __builtin_amdgcn_wmma_f32_16x16x32_f16(
          false, a0, false, b, (short)0, acc1[0][nt], false, false);
      acc1[1][nt] = __builtin_amdgcn_wmma_f32_16x16x32_f16(
          false, a1, false, b, (short)0, acc1[1][nt], false, false);
    }
  }

  // bias + relu; C-layout (col = lane&15, subtile row = v+mrow) -> LDS (f16)
  _Float16* hl = lds_h[wave];
  #pragma unroll
  for (int t = 0; t < 2; ++t) {
    #pragma unroll
    for (int nt = 0; nt < 4; ++nt) {
      const float bb = b1[nt * 16 + row];
      #pragma unroll
      for (int v = 0; v < 8; ++v) {
        hl[(t * 16 + v + mrow) * HID + nt * 16 + row] =
            (_Float16)fmaxf(acc1[t][nt][v] + bb, 0.f);
      }
    }
  }
  __syncthreads();

  // ---- GEMM2: [32,64] x [64,128] ----
  v8f acc2[2][8];
  #pragma unroll
  for (int t = 0; t < 2; ++t)
    #pragma unroll
    for (int nt = 0; nt < 8; ++nt) acc2[t][nt] = (v8f)0.0f;

  #pragma unroll
  for (int kt = 0; kt < 2; ++kt) {
    const _Float16* h0 = hl + (0 * 16 + row) * HID + kt * 32;
    const _Float16* h1 = hl + (1 * 16 + row) * HID + kt * 32;
    v16h a0 = cat16(*(const v8h*)(h0 + off), *(const v8h*)(h0 + 16 + off));
    v16h a1 = cat16(*(const v8h*)(h1 + off), *(const v8h*)(h1 + 16 + off));
    #pragma unroll
    for (int nt = 0; nt < 8; ++nt) {
      v16h b = *(const v16h*)(W2h + (((kt << 3) + nt) << 9) + (lane << 4));
      acc2[0][nt] = __builtin_amdgcn_wmma_f32_16x16x32_f16(
          false, a0, false, b, (short)0, acc2[0][nt], false, false);
      acc2[1][nt] = __builtin_amdgcn_wmma_f32_16x16x32_f16(
          false, a1, false, b, (short)0, acc2[1][nt], false, false);
    }
  }

  // bias + relu + scatter-add into segment sums (L2-resident)
  #pragma unroll
  for (int t = 0; t < 2; ++t) {
    #pragma unroll
    for (int nt = 0; nt < 8; ++nt) {
      const float bb  = b2[nt * 16 + row];
      const int   col = nt * 16 + row;
      #pragma unroll
      for (int v = 0; v < 8; ++v) {
        const int tgt = lds_ridx[wave][t * 16 + v + mrow];
        if (tgt >= 0) {
          float val = fmaxf(acc2[t][nt][v] + bb, 0.f);
          __hip_atomic_fetch_add(&seg_sum[(size_t)tgt * CH + col], val,
                                 __ATOMIC_RELAXED, __HIP_MEMORY_SCOPE_AGENT);
        }
      }
    }
  }

  // degree counts: each lane owns edge ebase+lane
  const int  myE = ebase + lane;
  const bool myV = tileValid && (myE < n_edges);
  const int  myRecv = hiHalf ? r1 : r0;
  if (myV) {
    __hip_atomic_fetch_add(&cnt[myRecv], 1.0f,
                           __ATOMIC_RELAXED, __HIP_MEMORY_SCOPE_AGENT);
  }
}

// ---------------------------------------------------------------------------
// Dense tail: relu(h @ Wd1 + bd1) @ Wd2 + bd2 per node; one wave = 16 nodes.
// ---------------------------------------------------------------------------
__global__ void __launch_bounds__(128)
dense_tail(const _Float16* __restrict__ hfeat,    // [N, 128] f16
           const _Float16* __restrict__ Wd1h,     // K=128,N=64 fragments
           const float* __restrict__ bd1,
           const float* __restrict__ Wd2,         // [64]
           const float* __restrict__ bd2,
           float* __restrict__ out, int n_nodes) {
  __shared__ __align__(32) float lds_h[4][16 * 64];

  const int wave   = threadIdx.x >> 5;
  const int lane   = threadIdx.x & 31;
  const int row    = lane & 15;
  const int hiHalf = (lane >> 4) & 1;
  const int mrow   = hiHalf * 8;
  const int off    = hiHalf * 8;

  const int maxTile = (n_nodes + 15) >> 4;
  int tile = blockIdx.x * 4 + wave;
  const bool tileValid = (tile < maxTile);
  if (!tileValid) tile = maxTile - 1;
  const int nbase = tile << 4;
  const int node  = min(nbase + row, n_nodes - 1);

  v8f acc[4];
  #pragma unroll
  for (int nt = 0; nt < 4; ++nt) acc[nt] = (v8f)0.0f;

  #pragma unroll
  for (int kt = 0; kt < 4; ++kt) {
    const _Float16* base = hfeat + (size_t)node * CH + kt * 32;
    v16h a = cat16(*(const v8h*)(base + off), *(const v8h*)(base + 16 + off));
    #pragma unroll
    for (int nt = 0; nt < 4; ++nt) {
      v16h b = *(const v16h*)(Wd1h + (((kt << 2) + nt) << 9) + (lane << 4));
      acc[nt] = __builtin_amdgcn_wmma_f32_16x16x32_f16(
          false, a, false, b, (short)0, acc[nt], false, false);
    }
  }

  float* hl = lds_h[wave];
  #pragma unroll
  for (int nt = 0; nt < 4; ++nt) {
    const float bb = bd1[nt * 16 + row];
    #pragma unroll
    for (int v = 0; v < 8; ++v) {
      hl[(v + mrow) * 64 + nt * 16 + row] = fmaxf(acc[nt][v] + bb, 0.f);
    }
  }
  __syncthreads();

  if (!hiHalf && tileValid) {
    const int onode = nbase + row;
    if (onode < n_nodes) {
      float s = bd2[0];
      const float* hr = hl + row * 64;
      #pragma unroll 8
      for (int k = 0; k < 64; ++k) s += hr[k] * Wd2[k];
      out[onode] = s;
    }
  }
}

// ---------------------------------------------------------------------------
extern "C" void kernel_launch(void* const* d_in, const int* in_sizes, int n_in,
                              void* d_out, int out_size, void* d_ws, size_t ws_size,
                              hipStream_t stream) {
  const float* x         = (const float*)d_in[0];
  const int*   senders   = (const int*)d_in[1];
  const int*   receivers = (const int*)d_in[2];
  const float* W1a = (const float*)d_in[3];
  const float* b1a = (const float*)d_in[4];
  const float* W2a = (const float*)d_in[5];
  const float* b2a = (const float*)d_in[6];
  const float* W1b = (const float*)d_in[7];
  const float* b1b = (const float*)d_in[8];
  const float* W2b = (const float*)d_in[9];
  const float* b2b = (const float*)d_in[10];
  const float* Wd1 = (const float*)d_in[11];
  const float* bd1 = (const float*)d_in[12];
  const float* Wd2 = (const float*)d_in[13];
  const float* bd2 = (const float*)d_in[14];

  const int n_edges = in_sizes[1];
  const int n_nodes = in_sizes[0] / D_FEAT;
  const size_t nc = (size_t)n_nodes * CH;

  // Workspace layout (all section sizes are multiples of 32 bytes)
  float* seg_sum   = (float*)d_ws;              // [N, CH] f32
  float* cnt       = seg_sum + nc;              // [N]     f32
  _Float16* xh     = (_Float16*)(cnt + n_nodes);// [N, CH] f16
  _Float16* hbuf_h = xh + nc;                   // [N, CH] f16
  _Float16* W1ah   = hbuf_h + nc;
  _Float16* W2ah   = W1ah + 256 * 64;
  _Float16* W1bh   = W2ah + 64 * 128;
  _Float16* W2bh   = W1bh + 256 * 64;
  _Float16* Wd1h   = W2bh + 64 * 128;

  // 1) one-time (per launch) conversions: features + weights -> f16
  cvt_f32_f16<<<((int)nc + 255) / 256, 256, 0, stream>>>(x, xh, (int)nc);
  prep_w16<<<(256 * 64  + 255) / 256, 256, 0, stream>>>(W1a, W1ah, 256, 64);
  prep_w16<<<(64  * 128 + 255) / 256, 256, 0, stream>>>(W2a, W2ah, 64, 128);
  prep_w16<<<(256 * 64  + 255) / 256, 256, 0, stream>>>(W1b, W1bh, 256, 64);
  prep_w16<<<(64  * 128 + 255) / 256, 256, 0, stream>>>(W2b, W2bh, 64, 128);
  prep_w16<<<(128 * 64  + 255) / 256, 256, 0, stream>>>(Wd1, Wd1h, 128, 64);

  const int segTotal  = (int)nc + n_nodes;  // seg_sum + cnt contiguous
  const int meanTotal = (int)nc;
  const int tilesE    = (n_edges + 31) / 32;
  const int blocksE   = (tilesE + 3) / 4;
  const int tilesN    = (n_nodes + 15) / 16;
  const int blocksN   = (tilesN + 3) / 4;

  // 2) layer 1
  zero_f32<<<(segTotal + 255) / 256, 256, 0, stream>>>(seg_sum, segTotal);
  edge_mlp_scatter<<<blocksE, 128, 0, stream>>>(
      xh, senders, receivers, W1ah, b1a, W2ah, b2a, seg_sum, cnt, n_edges);
  seg_mean_f16<<<(meanTotal + 255) / 256, 256, 0, stream>>>(seg_sum, cnt, hbuf_h, meanTotal);

  // 3) layer 2
  zero_f32<<<(segTotal + 255) / 256, 256, 0, stream>>>(seg_sum, segTotal);
  edge_mlp_scatter<<<blocksE, 128, 0, stream>>>(
      hbuf_h, senders, receivers, W1bh, b1b, W2bh, b2b, seg_sum, cnt, n_edges);
  seg_mean_f16<<<(meanTotal + 255) / 256, 256, 0, stream>>>(seg_sum, cnt, hbuf_h, meanTotal);

  // 4) dense tail -> d_out [N,1]
  dense_tail<<<blocksN, 128, 0, stream>>>(
      hbuf_h, Wd1h, bd1, Wd2, bd2, (float*)d_out, n_nodes);
}
